// ToyNet_46437186404650
// MI455X (gfx1250) — matrix operand order repaired and checked by reading, not compiled, not run on hardware
//
#include <hip/hip_runtime.h>
#include <hip/hip_bf16.h>
#include <cstdint>

typedef __attribute__((ext_vector_type(16))) _Float16 v16h;
typedef __attribute__((ext_vector_type(8)))  float    v8f;

#define IN_FEAT 128
#define HIDDEN  16
#define NCLS    40
#define NPAD    48   // NCLS padded to 3x16 for WMMA
#define KPAD    32   // HIDDEN padded to one K=32 WMMA step

// ---------------------------------------------------------------- utilities

__global__ void k_init_deg(float* __restrict__ deg, int n) {
  int i = blockIdx.x * blockDim.x + threadIdx.x;
  if (i < n) deg[i] = 1.0f;            // self-loop contributes 1 to degree
}

__global__ void k_zero(float* __restrict__ p, long long cnt) {
  long long i = (long long)blockIdx.x * blockDim.x + threadIdx.x;
  if (i < cnt) p[i] = 0.0f;
}

// Node ids are < 2^31, so only the low dword of each little-endian int64 matters.
__global__ void k_deg_edges(const long long* __restrict__ ei, long long E,
                            float* __restrict__ deg) {
  long long e = (long long)blockIdx.x * blockDim.x + threadIdx.x;
  if (e < E) {
    const int* ei32 = (const int*)ei;
    int d = ei32[2 * (E + e)];         // row 1 = dst, low dword
    atomicAdd(&deg[d], 1.0f);
  }
}

__global__ void k_rsqrt_inplace(float* __restrict__ deg, int n) {
  int i = blockIdx.x * blockDim.x + threadIdx.x;
  if (i < n) {
    float d = deg[i];
    deg[i] = (d > 0.0f) ? rsqrtf(d) : 0.0f;   // becomes dinv
  }
}

// ------------------------------------------------- GEMM1: h1 = x @ W1 (WMMA)
// One wave -> one 16x16 output tile.  K = 128 = 4 chunks of 32 (f16 WMMA).
// A (16x32 f16): m=lane&15, half=lane>>4; elements 0..7 hold K=kb+16*half+t,
//   elements 8..15 hold K=kb+16*half+8+t.
// B (32x16 f16): N=lane&15; element e holds K=kb+16*half+e.
// C (16x16 f32): VGPR r -> M = r + 8*half, N = lane&15.

__global__ void k_gemm1_wmma(const float* __restrict__ x,
                             const float* __restrict__ W1,
                             float* __restrict__ h1, int tiles) {
  __shared__ __align__(32) _Float16 w1t[HIDDEN * IN_FEAT];  // [n][k] transposed
  for (int i = threadIdx.x; i < HIDDEN * IN_FEAT; i += blockDim.x) {
    int nn = i >> 7;           // output col
    int kk = i & 127;          // k
    w1t[i] = (_Float16)W1[kk * HIDDEN + nn];
  }
  __syncthreads();

  int lane = threadIdx.x & 31;
  int wv   = threadIdx.x >> 5;
  int tile = blockIdx.x * 4 + wv;
  if (tile >= tiles) return;                 // wave-uniform: EXEC all-1s inside

  int m    = lane & 15;
  int half = lane >> 4;
  const float* xrow = x + (long long)(tile * 16 + m) * IN_FEAT;

  v8f acc = {};
#pragma unroll
  for (int kc = 0; kc < 4; ++kc) {
    int kb = kc * 32;
    const float4* p = (const float4*)(xrow + kb + 16 * half);
    const float4* q = (const float4*)(xrow + kb + 16 * half + 8);
    float4 a0 = p[0], a1 = p[1];
    float4 a2 = q[0], a3 = q[1];
    float af[16] = {a0.x, a0.y, a0.z, a0.w, a1.x, a1.y, a1.z, a1.w,
                    a2.x, a2.y, a2.z, a2.w, a3.x, a3.y, a3.z, a3.w};
    v16h a;
#pragma unroll
    for (int t = 0; t < 16; ++t) a[t] = (_Float16)af[t];

    const v16h* bp = (const v16h*)&w1t[m * IN_FEAT + kb + 16 * half];
    v16h b = *bp;

    acc = __builtin_amdgcn_wmma_f32_16x16x32_f16(
        false, a, false, b, (short)0, acc, false, false);
  }

  float* o = h1 + (long long)tile * 16 * HIDDEN;
#pragma unroll
  for (int r = 0; r < 8; ++r)
    o[(r + 8 * half) * HIDDEN + m] = acc[r];
}

// scalar fallback for a partial last tile (unused when n % 16 == 0)
__global__ void k_gemm1_tail(const float* __restrict__ x,
                             const float* __restrict__ W1,
                             float* __restrict__ h1, int n, int rem) {
  int idx = blockIdx.x * blockDim.x + threadIdx.x;
  if (idx >= rem * HIDDEN) return;
  int r = n - rem + idx / HIDDEN;
  int f = idx % HIDDEN;
  float s = 0.0f;
  for (int k = 0; k < IN_FEAT; ++k)
    s += x[(long long)r * IN_FEAT + k] * W1[k * HIDDEN + f];
  h1[(long long)r * HIDDEN + f] = s;
}

// ------------------------------------------------- edge scatter (both layers)
// Thread per (edge, 4-feature quad): one b128 payload load, 4 f32 atomics.

template <int F>
__global__ void k_scatter(const long long* __restrict__ ei, long long E,
                          const float* __restrict__ dinv,
                          const float* __restrict__ h,
                          float* __restrict__ agg) {
  constexpr int Q = F / 4;
  long long idx = (long long)blockIdx.x * blockDim.x + threadIdx.x;
  if (idx >= E * Q) return;
  long long e = idx / Q;
  int q = (int)(idx - e * Q);
  const int* ei32 = (const int*)ei;          // low dwords of int64 ids
  int s = ei32[2 * e];
  int d = ei32[2 * (E + e)];
  float norm = dinv[s] * dinv[d];
  float4 hv = *(const float4*)(h + (long long)s * F + 4 * q);
  float* ap = agg + (long long)d * F + 4 * q;
  atomicAdd(ap + 0, hv.x * norm);
  atomicAdd(ap + 1, hv.y * norm);
  atomicAdd(ap + 2, hv.z * norm);
  atomicAdd(ap + 3, hv.w * norm);
}

// self-loop term + bias + ReLU, in place on agg1
__global__ void k_selfloop_bias_relu(const float* __restrict__ h1,
                                     float* __restrict__ agg1,
                                     const float* __restrict__ dinv,
                                     const float* __restrict__ b1, int n) {
  long long idx = (long long)blockIdx.x * blockDim.x + threadIdx.x;
  if (idx >= (long long)n * HIDDEN) return;
  int i = (int)(idx / HIDDEN);
  int f = (int)(idx - (long long)i * HIDDEN);
  float di = dinv[i];
  float v = agg1[idx] + h1[idx] * di * di + b1[f];
  agg1[idx] = v > 0.0f ? v : 0.0f;
}

// ------------------------------------------------- GEMM2: h2 = hrelu @ W2 (WMMA)
// N padded 40->48 (3 column tiles), K padded 16->32 (upper A half = 0).
// One wave -> 16 rows x 48 cols with 3 WMMAs sharing the same A fragment.

__global__ void k_gemm2_wmma(const float* __restrict__ h,
                             const float* __restrict__ W2,
                             float* __restrict__ h2, int tiles) {
  __shared__ __align__(32) _Float16 w2t[NPAD * KPAD];   // [n_pad][k_pad] transposed
  for (int i = threadIdx.x; i < NPAD * KPAD; i += blockDim.x) {
    int nn = i >> 5;           // padded output col
    int kk = i & 31;           // padded k
    w2t[i] = (nn < NCLS && kk < HIDDEN) ? (_Float16)W2[kk * NCLS + nn]
                                        : (_Float16)0.0f;
  }
  __syncthreads();

  int lane = threadIdx.x & 31;
  int wv   = threadIdx.x >> 5;
  int tile = blockIdx.x * 4 + wv;
  if (tile >= tiles) return;

  int m    = lane & 15;
  int half = lane >> 4;
  const float* hrow = h + (long long)(tile * 16 + m) * HIDDEN;

  // A fragment: elements 0..7 = K = 8*half + t (< 16, valid); 8..15 = K>=16 -> 0
  const float4* p = (const float4*)(hrow + 8 * half);
  float4 a0 = p[0], a1 = p[1];
  float af[8] = {a0.x, a0.y, a0.z, a0.w, a1.x, a1.y, a1.z, a1.w};
  v16h a;
#pragma unroll
  for (int t = 0; t < 8; ++t) {
    a[t]     = (_Float16)af[t];
    a[8 + t] = (_Float16)0.0f;
  }

  v8f acc[3];
#pragma unroll
  for (int ct = 0; ct < 3; ++ct) {
    const _Float16* bq = &w2t[(ct * 16 + m) * KPAD + 8 * half];
    v16h b;
#pragma unroll
    for (int t = 0; t < 8; ++t) {
      b[t]     = bq[t];
      b[8 + t] = bq[16 + t];
    }
    v8f c = {};
    acc[ct] = __builtin_amdgcn_wmma_f32_16x16x32_f16(
        false, a, false, b, (short)0, c, false, false);
  }

  float* o = h2 + (long long)tile * 16 * NCLS;
#pragma unroll
  for (int ct = 0; ct < 3; ++ct) {
    int col = ct * 16 + m;
    if (col < NCLS) {
#pragma unroll
      for (int r = 0; r < 8; ++r)
        o[(r + 8 * half) * NCLS + col] = acc[ct][r];
    }
  }
}

// scalar fallback for a partial last tile (unused when n % 16 == 0)
__global__ void k_gemm2_tail(const float* __restrict__ h,
                             const float* __restrict__ W2,
                             float* __restrict__ h2, int n, int rem) {
  int idx = blockIdx.x * blockDim.x + threadIdx.x;
  if (idx >= rem * NCLS) return;
  int r = n - rem + idx / NCLS;
  int c = idx % NCLS;
  const float* hr = h + (long long)r * HIDDEN;
  float s = 0.0f;
#pragma unroll
  for (int f = 0; f < HIDDEN; ++f) s += hr[f] * W2[f * NCLS + c];
  h2[(long long)r * NCLS + c] = s;
}

// self-loop term + bias + log_softmax, in place on out
__global__ void k_final(float* __restrict__ out, const float* __restrict__ h2,
                        const float* __restrict__ dinv,
                        const float* __restrict__ b2, int n) {
  int i = blockIdx.x * blockDim.x + threadIdx.x;
  if (i >= n) return;
  float di = dinv[i];
  float d2 = di * di;
  long long base = (long long)i * NCLS;
  float mx = -1e30f;
  for (int c = 0; c < NCLS; ++c) {
    float v = out[base + c] + h2[base + c] * d2 + b2[c];
    mx = fmaxf(mx, v);
  }
  float s = 0.0f;
  for (int c = 0; c < NCLS; ++c) {
    float v = out[base + c] + h2[base + c] * d2 + b2[c];
    s += __expf(v - mx);
  }
  float l = mx + __logf(s);
  for (int c = 0; c < NCLS; ++c) {
    float v = out[base + c] + h2[base + c] * d2 + b2[c];
    out[base + c] = v - l;
  }
}

// ---------------------------------------------------------------- launcher

extern "C" void kernel_launch(void* const* d_in, const int* in_sizes, int n_in,
                              void* d_out, int out_size, void* d_ws, size_t ws_size,
                              hipStream_t stream) {
  const float*     x  = (const float*)d_in[0];
  const long long* ei = (const long long*)d_in[1];   // int64 edge_index [2, E]
  const float*     W1 = (const float*)d_in[2];
  const float*     b1 = (const float*)d_in[3];
  const float*     W2 = (const float*)d_in[4];
  const float*     b2 = (const float*)d_in[5];
  float*           out = (float*)d_out;

  const int       n = in_sizes[0] / IN_FEAT;
  const long long E = (long long)in_sizes[1] / 2;

  float* ws   = (float*)d_ws;
  float* dinv = ws;                       // n         (deg -> dinv in place)
  float* h1   = ws + (long long)n;        // 16n
  float* agg1 = ws + 17LL * n;            // 16n       (becomes relu'd hidden)
  float* h2   = ws + 33LL * n;            // 40n

  const int B = 256;
  auto blks = [](long long c, int b) { return (unsigned)((c + b - 1) / b); };

  k_init_deg<<<blks(n, B), B, 0, stream>>>(dinv, n);
  k_zero<<<blks(16LL * n, B), B, 0, stream>>>(agg1, 16LL * n);
  k_zero<<<blks(40LL * n, B), B, 0, stream>>>(out, 40LL * n);

  k_deg_edges<<<blks(E, B), B, 0, stream>>>(ei, E, dinv);
  k_rsqrt_inplace<<<blks(n, B), B, 0, stream>>>(dinv, n);

  int tiles = n / 16;
  int rem   = n & 15;
  if (tiles > 0)
    k_gemm1_wmma<<<blks(tiles, 4), 128, 0, stream>>>(x, W1, h1, tiles);
  if (rem)
    k_gemm1_tail<<<1, 256, 0, stream>>>(x, W1, h1, n, rem);

  k_scatter<HIDDEN><<<blks(E * (HIDDEN / 4), B), B, 0, stream>>>(ei, E, dinv, h1, agg1);
  k_selfloop_bias_relu<<<blks(16LL * n, B), B, 0, stream>>>(h1, agg1, dinv, b1, n);

  if (tiles > 0)
    k_gemm2_wmma<<<blks(tiles, 4), 128, 0, stream>>>(agg1, W2, h2, tiles);
  if (rem)
    k_gemm2_tail<<<1, 256, 0, stream>>>(agg1, W2, h2, n, rem);

  k_scatter<NCLS><<<blks(E * (NCLS / 4), B), B, 0, stream>>>(ei, E, dinv, h2, out);
  k_final<<<blks(n, B), B, 0, stream>>>(out, h2, dinv, b2, n);
}
